// SparseMoE_75290776699500
// MI455X (gfx1250) — matrix operand order repaired and checked by reading, not compiled, not run on hardware
//
#include <hip/hip_runtime.h>
#include <math.h>

// ---------------- problem constants ----------------
#define Dd 1024   // in/out features
#define Hh 4096   // hidden features
#define Ee 8      // experts
#define Tt 8192   // tokens (8*1024)
// ---------------- GEMM tiling ----------------------
#define BM 128
#define BN 128
#define BK 32

typedef __attribute__((ext_vector_type(16))) _Float16 v16h;
typedef __attribute__((ext_vector_type(8)))  float    v8f;

union Frag16 { v16h v; uint4 q[2]; };

static __device__ __forceinline__ unsigned short f2h(float f) {
  union { _Float16 h; unsigned short u; } c; c.h = (_Float16)f; return c.u;
}

// 16B global->LDS async copy (gfx1250 GLOBAL_LOAD_ASYNC_TO_LDS_B128, ASYNCcnt).
// VDST = per-lane LDS byte offset (low 32 bits of the generic shared pointer;
// HW adds LDS_BASE), VADDR = 64-bit global address, no SADDR ("off").
static __device__ __forceinline__ void async_cp16(const unsigned short* gsrc,
                                                  unsigned short* ldst) {
  unsigned lds_off = (unsigned)(unsigned long long)(uintptr_t)ldst;
  unsigned long long gaddr = (unsigned long long)(uintptr_t)gsrc;
  asm volatile("global_load_async_to_lds_b128 %0, %1, off"
               :: "v"(lds_off), "v"(gaddr) : "memory");
}
static __device__ __forceinline__ void wait_async0() {
  asm volatile("s_wait_asynccnt 0x0" ::: "memory");
}

// ------------------------------------------------------------------
// Kernel 0: zero output accumulator + per-expert counters
// ------------------------------------------------------------------
__global__ void zero_init_kernel(float* __restrict__ out, int* __restrict__ counts) {
  size_t i = (size_t)blockIdx.x * blockDim.x + threadIdx.x;
  float4 z; z.x = 0.f; z.y = 0.f; z.z = 0.f; z.w = 0.f;
  ((float4*)out)[i] = z;
  if (blockIdx.x == 0 && threadIdx.x < Ee) counts[threadIdx.x] = 0;
}

// ------------------------------------------------------------------
// Kernel 1: x fp32 -> f16 once (GEMM A-tiles then stream as f16)
// ------------------------------------------------------------------
__global__ void cvt_x_kernel(const float* __restrict__ x, unsigned short* __restrict__ xh) {
  size_t i = (size_t)blockIdx.x * blockDim.x + threadIdx.x;
  float4 f = ((const float4*)x)[i];
  union { unsigned short s[4]; uint2 v; } p;
  p.s[0] = f2h(f.x); p.s[1] = f2h(f.y); p.s[2] = f2h(f.z); p.s[3] = f2h(f.w);
  ((uint2*)xh)[i] = p.v;
}

// ------------------------------------------------------------------
// Kernel 2: router. One wave32 per token: logits, softmax top-2,
// normalized weights, atomic-append into per-expert gather lists.
// ------------------------------------------------------------------
__global__ void router_kernel(const float* __restrict__ x,
                              const float* __restrict__ rw,   // [Dd][Ee]
                              int* __restrict__ counts,
                              int* __restrict__ ids,          // [Ee][Tt]
                              float* __restrict__ wts) {      // [Ee][Tt]
  const int wave = threadIdx.x >> 5;
  const int lane = threadIdx.x & 31;
  const int t = blockIdx.x * 8 + wave;
  const float* xr = x + (size_t)t * Dd;

  float acc[Ee];
#pragma unroll
  for (int e = 0; e < Ee; ++e) acc[e] = 0.f;

  for (int j = 0; j < Dd / 32; ++j) {
    const int d = lane + j * 32;
    const float xv = xr[d];
    const float4 r0 = *(const float4*)(rw + (size_t)d * Ee);
    const float4 r1 = *(const float4*)(rw + (size_t)d * Ee + 4);
    acc[0] += xv * r0.x; acc[1] += xv * r0.y; acc[2] += xv * r0.z; acc[3] += xv * r0.w;
    acc[4] += xv * r1.x; acc[5] += xv * r1.y; acc[6] += xv * r1.z; acc[7] += xv * r1.w;
  }
#pragma unroll
  for (int off = 16; off > 0; off >>= 1)
#pragma unroll
    for (int e = 0; e < Ee; ++e) acc[e] += __shfl_xor(acc[e], off, 32);

  if (lane == 0) {
    float mx = acc[0];
#pragma unroll
    for (int e = 1; e < Ee; ++e) mx = fmaxf(mx, acc[e]);
    float p[Ee];
#pragma unroll
    for (int e = 0; e < Ee; ++e) p[e] = __expf(acc[e] - mx);
    int i1 = 0;
#pragma unroll
    for (int e = 1; e < Ee; ++e) if (p[e] > p[i1]) i1 = e;   // ties -> lowest index
    int i2 = (i1 == 0) ? 1 : 0;
#pragma unroll
    for (int e = 0; e < Ee; ++e) if (e != i1 && p[e] > p[i2]) i2 = e;
    const float s = p[i1] + p[i2];
    const int pos1 = atomicAdd(&counts[i1], 1);
    ids[i1 * Tt + pos1] = t;  wts[i1 * Tt + pos1] = p[i1] / s;
    const int pos2 = atomicAdd(&counts[i2], 1);
    ids[i2 * Tt + pos2] = t;  wts[i2 * Tt + pos2] = p[i2] / s;
  }
}

// ==================================================================
// Double-buffered, async-staged WMMA mainloop (shared by both GEMMs)
//   As[b][M][K]  : f16, filled by global_load_async_to_lds_b128
//   Bs[b][N][K]  : f16 (transposed), fp32->f16 converted via regs
// 256 threads = 8 waves; wave grid 2(M) x 4(N); 8 C-tiles per wave.
// Fragment addressing per ISA 7.12.2 wave32 layouts:
//   A lane m: halves 0-7 -> K {0|8}+i, halves 8-15 -> K {16|24}+i
//   B lane n: halves 0-7 -> K {0|8}+i, halves 8-15 -> K {16|24}+i
//   C vgpr g -> row = g + 8*(lane>=16), col = lane&15
// ==================================================================
struct TileCtx {
  int lane, wm, wn, lhi, l15;
  int am, ah, bn, bk;
};

static __device__ __forceinline__ TileCtx make_ctx(int tid) {
  TileCtx c;
  c.lane = tid & 31;
  const int wv = tid >> 5;
  c.wm  = wv & 1;
  c.wn  = wv >> 1;
  c.lhi = (c.lane >> 4) & 1;
  c.l15 = c.lane & 15;
  c.am  = tid >> 1;            // A row (2 threads/row)
  c.ah  = (tid & 1) * 16;      // halves offset within BK
  c.bn  = (tid & 63) * 2;      // 2 B columns / thread
  c.bk  = (tid >> 6) * 8;      // 8 B k-rows / thread
  return c;
}

// Load 8 float2 of a B tile slice into registers (next-step prefetch).
static __device__ __forceinline__ void load_bregs(const float* __restrict__ wB,
                                                  int ldb, int k0, int n0,
                                                  const TileCtx& c, float2* br) {
#pragma unroll
  for (int kk = 0; kk < 8; ++kk)
    br[kk] = *(const float2*)(wB + (size_t)(k0 + c.bk + kk) * ldb + n0 + c.bn);
}

// Convert register-held B slice and store transposed [N][K] into LDS.
static __device__ __forceinline__ void store_btile(unsigned short* Bsb,
                                                   const TileCtx& c, const float2* br) {
  union { unsigned short s[8]; uint4 v; } c0, c1;
#pragma unroll
  for (int kk = 0; kk < 8; ++kk) { c0.s[kk] = f2h(br[kk].x); c1.s[kk] = f2h(br[kk].y); }
  *(uint4*)&Bsb[(c.bn)     * BK + c.bk] = c0.v;
  *(uint4*)&Bsb[(c.bn + 1) * BK + c.bk] = c1.v;
}

// Load fragments from LDS and run the 8 WMMAs.
static __device__ __forceinline__ void mma_step(const unsigned short* Asb,
                                                const unsigned short* Bsb,
                                                const TileCtx& c, v8f* acc) {
  Frag16 a[4], b[2];
#pragma unroll
  for (int mt = 0; mt < 4; ++mt) {
    const unsigned short* p = &Asb[(c.wm * 64 + mt * 16 + c.l15) * BK];
    a[mt].q[0] = *(const uint4*)(p + c.lhi * 8);
    a[mt].q[1] = *(const uint4*)(p + c.lhi * 8 + 16);
  }
#pragma unroll
  for (int nt = 0; nt < 2; ++nt) {
    const unsigned short* p = &Bsb[(c.wn * 32 + nt * 16 + c.l15) * BK + c.lhi * 8];
    b[nt].q[0] = *(const uint4*)(p);
    b[nt].q[1] = *(const uint4*)(p + 16);
  }
#pragma unroll
  for (int mt = 0; mt < 4; ++mt)
#pragma unroll
    for (int nt = 0; nt < 2; ++nt)
      acc[mt * 2 + nt] = __builtin_amdgcn_wmma_f32_16x16x32_f16(
          false, a[mt].v, false, b[nt].v, (short)0, acc[mt * 2 + nt], false, false);
}

// ------------------------------------------------------------------
// GEMM1 per expert: h = gelu( gather(x_f16) @ w1[e] + b1[e] ) -> f16
// ------------------------------------------------------------------
__global__ __launch_bounds__(256) void gemm1_kernel(
    const unsigned short* __restrict__ xh,   // [Tt][Dd] f16
    const float* __restrict__ w1e,           // [Dd][Hh]
    const float* __restrict__ b1e,           // [Hh]
    const int* __restrict__ cnt_p,
    const int* __restrict__ ids_e,           // [Tt]
    unsigned short* __restrict__ hh) {       // [Tt][Hh] f16
  const int cnt = *cnt_p;
  const int m0 = blockIdx.y * BM;
  if (m0 >= cnt) return;
  const int n0 = blockIdx.x * BN;

  __shared__ unsigned short As[2][BM * BK];
  __shared__ unsigned short Bs[2][BN * BK];
  __shared__ int sId[BM];

  const int tid = threadIdx.x;
  if (tid < BM) {
    const int r = m0 + tid;
    sId[tid] = (r < cnt) ? ids_e[r] : 0;   // pad with token 0 (weight 0 downstream)
  }
  __syncthreads();

  const TileCtx c = make_ctx(tid);
  const unsigned short* asrc = xh + (size_t)sId[c.am] * Dd + c.ah;
  unsigned short* aDst[2] = { &As[0][c.am * BK + c.ah], &As[1][c.am * BK + c.ah] };

  v8f acc[8];
#pragma unroll
  for (int nt = 0; nt < 2; ++nt) {
    const float bias = b1e[n0 + c.wn * 32 + nt * 16 + c.l15];
#pragma unroll
    for (int mt = 0; mt < 4; ++mt)
#pragma unroll
      for (int g = 0; g < 8; ++g) acc[mt * 2 + nt][g] = bias;
  }

  // -------- pipeline prologue: stage k=0 --------
  async_cp16(asrc, aDst[0]);
  async_cp16(asrc + 8, aDst[0] + 8);
  float2 br[8];
  load_bregs(w1e, Hh, 0, n0, c, br);

  int buf = 0;
  for (int k0 = 0; k0 < Dd; k0 += BK) {
    store_btile(&Bs[buf][0], c, br);
    wait_async0();               // own async A-tile copies landed
    __syncthreads();             // everyone's A+B tiles for `buf` visible
    if (k0 + BK < Dd) {          // stage k0+BK into the other buffer
      async_cp16(asrc + k0 + BK, aDst[buf ^ 1]);
      async_cp16(asrc + k0 + BK + 8, aDst[buf ^ 1] + 8);
      load_bregs(w1e, Hh, k0 + BK, n0, c, br);
    }
    mma_step(&As[buf][0], &Bs[buf][0], c, acc);
    buf ^= 1;
  }

  // ---- epilogue: exact GELU, store h as f16 ----
#pragma unroll
  for (int mt = 0; mt < 4; ++mt)
#pragma unroll
    for (int nt = 0; nt < 2; ++nt)
#pragma unroll
      for (int g = 0; g < 8; ++g) {
        const int row = m0 + c.wm * 64 + mt * 16 + g + c.lhi * 8;
        const int col = n0 + c.wn * 32 + nt * 16 + c.l15;
        const float v = acc[mt * 2 + nt][g];
        const float gel = 0.5f * v * (1.0f + erff(v * 0.70710678118654752f));
        hh[(size_t)row * Hh + col] = f2h(gel);
      }
}

// ------------------------------------------------------------------
// GEMM2 per expert: y = h @ w2[e] + b2[e]; out[token] += weight * y
// ------------------------------------------------------------------
__global__ __launch_bounds__(256) void gemm2_kernel(
    const unsigned short* __restrict__ hh,   // [Tt][Hh] f16
    const float* __restrict__ w2e,           // [Hh][Dd]
    const float* __restrict__ b2e,           // [Dd]
    const int* __restrict__ cnt_p,
    const int* __restrict__ ids_e,
    const float* __restrict__ wts_e,
    float* __restrict__ out) {               // [Tt][Dd]
  const int cnt = *cnt_p;
  const int m0 = blockIdx.y * BM;
  if (m0 >= cnt) return;
  const int n0 = blockIdx.x * BN;

  __shared__ unsigned short As[2][BM * BK];
  __shared__ unsigned short Bs[2][BN * BK];
  __shared__ int   sId[BM];
  __shared__ float sW[BM];

  const int tid = threadIdx.x;
  if (tid < BM) {
    const int r = m0 + tid;
    const bool ok = (r < cnt);
    sId[tid] = ok ? ids_e[r] : 0;
    sW[tid]  = ok ? wts_e[r] : 0.f;
  }
  __syncthreads();

  const TileCtx c = make_ctx(tid);
  const unsigned short* asrc = hh + (size_t)(m0 + c.am) * Hh + c.ah;
  unsigned short* aDst[2] = { &As[0][c.am * BK + c.ah], &As[1][c.am * BK + c.ah] };

  v8f acc[8];
#pragma unroll
  for (int nt = 0; nt < 2; ++nt) {
    const float bias = b2e[n0 + c.wn * 32 + nt * 16 + c.l15];
#pragma unroll
    for (int mt = 0; mt < 4; ++mt)
#pragma unroll
      for (int g = 0; g < 8; ++g) acc[mt * 2 + nt][g] = bias;
  }

  async_cp16(asrc, aDst[0]);
  async_cp16(asrc + 8, aDst[0] + 8);
  float2 br[8];
  load_bregs(w2e, Dd, 0, n0, c, br);

  int buf = 0;
  for (int k0 = 0; k0 < Hh; k0 += BK) {
    store_btile(&Bs[buf][0], c, br);
    wait_async0();
    __syncthreads();
    if (k0 + BK < Hh) {
      async_cp16(asrc + k0 + BK, aDst[buf ^ 1]);
      async_cp16(asrc + k0 + BK + 8, aDst[buf ^ 1] + 8);
      load_bregs(w2e, Dd, k0 + BK, n0, c, br);
    }
    mma_step(&As[buf][0], &Bs[buf][0], c, acc);
    buf ^= 1;
  }

  // ---- epilogue: scale by routing weight, accumulate into out ----
#pragma unroll
  for (int mt = 0; mt < 4; ++mt)
#pragma unroll
    for (int nt = 0; nt < 2; ++nt)
#pragma unroll
      for (int g = 0; g < 8; ++g) {
        const int rowl = c.wm * 64 + mt * 16 + g + c.lhi * 8;
        const float w = sW[rowl];
        if (w > 0.f) {
          const int tok = sId[rowl];
          const int col = n0 + c.wn * 32 + nt * 16 + c.l15;
          unsafeAtomicAdd(&out[(size_t)tok * Dd + col], w * acc[mt * 2 + nt][g]);
        }
      }
}

// ------------------------------------------------------------------
// Host-side launch sequence (all on `stream`, graph-capture safe)
// Workspace: counts(256B) | ids 256KB | wts 256KB | x_f16 16MB | h_f16 64MB
// ------------------------------------------------------------------
extern "C" void kernel_launch(void* const* d_in, const int* in_sizes, int n_in,
                              void* d_out, int out_size, void* d_ws, size_t ws_size,
                              hipStream_t stream) {
  const float* x  = (const float*)d_in[0];
  const float* rw = (const float*)d_in[1];
  const float* w1 = (const float*)d_in[2];
  const float* b1 = (const float*)d_in[3];
  const float* w2 = (const float*)d_in[4];
  const float* b2 = (const float*)d_in[5];
  float* out = (float*)d_out;

  char* ws = (char*)d_ws;
  int*            counts = (int*)ws;                                   // 8 ints
  int*            ids    = (int*)(ws + 256);                           // [E][T]
  float*          wts    = (float*)(ws + 256 + (size_t)Ee * Tt * 4);   // [E][T]
  unsigned short* xh     = (unsigned short*)(ws + 256 + 2ull * Ee * Tt * 4);
  unsigned short* hh     = xh + (size_t)Tt * Dd;                       // [T][H] f16

  const int vec_blocks = (Tt * Dd / 4) / 256;  // 8192
  zero_init_kernel<<<vec_blocks, 256, 0, stream>>>(out, counts);
  cvt_x_kernel<<<vec_blocks, 256, 0, stream>>>(x, xh);
  router_kernel<<<Tt / 8, 256, 0, stream>>>(x, rw, counts, ids, wts);

  for (int e = 0; e < Ee; ++e) {
    gemm1_kernel<<<dim3(Hh / BN, Tt / BM), 256, 0, stream>>>(
        xh,
        w1 + (size_t)e * Dd * Hh,
        b1 + (size_t)e * Hh,
        counts + e,
        ids + (size_t)e * Tt,
        hh);
    gemm2_kernel<<<dim3(Dd / BN, Tt / BM), 256, 0, stream>>>(
        hh,
        w2 + (size_t)e * Hh * Dd,
        b2 + (size_t)e * Dd,
        counts + e,
        ids + (size_t)e * Tt,
        wts + (size_t)e * Tt,
        out);
  }
}
